// MultiHeadTopologicalAttention_30133490549637
// MI455X (gfx1250) — matrix operand order repaired and checked by reading, not compiled, not run on hardware
//
#include <hip/hip_runtime.h>

// ---------------------------------------------------------------------------
// MultiHeadTopologicalAttention for MI455X (gfx1250, wave32, WMMA + TDM)
// ---------------------------------------------------------------------------
typedef __attribute__((ext_vector_type(16))) __bf16       v16bf;
typedef __attribute__((ext_vector_type(8)))  __bf16       v8bf;
typedef __attribute__((ext_vector_type(8)))  float        v8f;
typedef __attribute__((ext_vector_type(4)))  unsigned int v4u;
typedef __attribute__((ext_vector_type(4)))  int          v4i;
typedef __attribute__((ext_vector_type(8)))  int          v8i;

#define B_   2
#define S_   2048
#define H_   8
#define D_   64
#define DM_  512   // N_HEAD * D_ATTN == D_MODEL == 512

static __device__ __forceinline__ v8f wmma_bf16(v16bf a, v16bf b, v8f c) {
  // D = A(16x32 bf16) x B(32x16 bf16) + C(16x16 f32)
  return __builtin_amdgcn_wmma_f32_16x16x32_bf16(false, a, false, b,
                                                 (short)0, c, false, false);
}

// contiguous 16 bf16 (32B)
static __device__ __forceinline__ v16bf ld16(const __bf16* p) {
  return *reinterpret_cast<const v16bf*>(p);
}
// A-fragment pattern: 8 contiguous + 8 contiguous (callers pass hi = lo + 8: k = 16*grp + i)
static __device__ __forceinline__ v16bf ld8x2(const __bf16* lo, const __bf16* hi) {
  v8bf a = *reinterpret_cast<const v8bf*>(lo);
  v8bf b = *reinterpret_cast<const v8bf*>(hi);
  return __builtin_shufflevector(a, b, 0,1,2,3,4,5,6,7,8,9,10,11,12,13,14,15);
}

// generic pointer -> LDS byte offset (low 32 bits of the LDS aperture address)
static __device__ __forceinline__ unsigned lds_off(const void* p) {
  return (unsigned)(unsigned long long)(uintptr_t)p;
}

// ---------------------------------------------------------------------------
// Tensor Data Mover: 2-D tile load, bf16 elements (data_size = 2 bytes).
// D# bitfields per CDNA5 ISA:
//   group0: count=1 | lds_addr | global_addr[56:0] | type=2
//   group1: data_size=1(2B), tensor_dim0/1, tile_dim0/1, tensor_dim0_stride
//   groups 2/3: zero (2-D tensor, tile_dim2 = 0)
// dims/strides in element units, addresses in bytes. Tracked by TENSORcnt.
// ---------------------------------------------------------------------------
static __device__ __forceinline__ void tdm_load_2d_bf16(
    unsigned ldsoff, const void* g,
    unsigned tensor_d0, unsigned tensor_d1,
    unsigned tile_d0, unsigned tile_d1,
    unsigned long long stride0)
{
  const unsigned long long ga = (unsigned long long)(uintptr_t)g;
  v4u g0;
  g0[0] = 1u;                                                  // count=1 (valid)
  g0[1] = ldsoff;                                              // lds_addr
  g0[2] = (unsigned)(ga & 0xffffffffull);                      // global_addr lo
  g0[3] = (unsigned)((ga >> 32) & 0x1ffffffull) | (2u << 30);  // addr hi | type=2
  v8i g1;
  g1[0] = 1 << 16;                                             // data_size=2B, mask=0
  g1[1] = (int)((tensor_d0 & 0xffffu) << 16);                  // [63:48] td0 lo
  g1[2] = (int)(((tensor_d0 >> 16) & 0xffffu)
              | ((tensor_d1 & 0xffffu) << 16));                // td0 hi | td1 lo
  g1[3] = (int)(((tensor_d1 >> 16) & 0xffffu)
              | ((tile_d0 & 0xffffu) << 16));                  // td1 hi | tile_d0
  g1[4] = (int)(tile_d1 & 0xffffu);                            // tile_d1, tile_d2=0
  g1[5] = (int)(stride0 & 0xffffffffull);                      // stride0 lo
  g1[6] = (int)((stride0 >> 32) & 0xffffull);                  // stride0 hi
  g1[7] = 0;
  const v4i z4 = {0, 0, 0, 0};
#if defined(__clang_major__) && (__clang_major__ >= 23)
  const v8i z8 = {0, 0, 0, 0, 0, 0, 0, 0};
  __builtin_amdgcn_tensor_load_to_lds(g0, g1, z4, z4, z8, 0);
#else
  __builtin_amdgcn_tensor_load_to_lds(g0, g1, z4, z4, 0);
#endif
}

// ---------------------------------------------------------------------------
// Kernel 1: fused QKV projection.  out = x @ W^T + b, f32 in -> bf16 out.
// Q,K stored [b,h,s,d]; V stored transposed [b,h,d,s] for contiguous PV B-frags.
// ---------------------------------------------------------------------------
__global__ __launch_bounds__(256) void qkv_proj_kernel(
    const float* __restrict__ x,
    const float* __restrict__ wq, const float* __restrict__ bq,
    const float* __restrict__ wk, const float* __restrict__ bk,
    const float* __restrict__ wv, const float* __restrict__ bv,
    __bf16* __restrict__ qb, __bf16* __restrict__ kbm, __bf16* __restrict__ vt)
{
  const int wave = threadIdx.x >> 5, lane = threadIdx.x & 31;
  const int grp = lane >> 4, ln = lane & 15;
  const int tile = blockIdx.x * 8 + wave;      // 0 .. 24575
  const int rowTile = tile / 96;               // 0 .. 255  (4096 tokens / 16)
  const int colTile = tile % 96;               // 0 .. 95   (3 * 512 / 16)
  const int mat = colTile >> 5;                // 0=Q 1=K 2=V
  const int ct  = colTile & 31;                // col tile within matrix
  const float* __restrict__ w  = (mat == 0) ? wq : (mat == 1) ? wk : wv;
  const float* __restrict__ bb = (mat == 0) ? bq : (mat == 1) ? bk : bv;
  const int arow = rowTile * 16 + ln;          // A-matrix row (M = lane%16)
  const int bcol = ct * 16 + ln;               // B-matrix col (N = lane%16)

  v8f acc = {};
  for (int kt = 0; kt < DM_; kt += 32) {
    v16bf a, bf;
    const float* ap = x + (size_t)arow * DM_ + kt;
    const float* bp = w + (size_t)bcol * DM_ + kt;
#pragma unroll
    for (int i = 0; i < 8; ++i) {              // A: K = grp*16 + i (contiguous)
      a[i]     = (__bf16)ap[grp * 16 + i];
      a[i + 8] = (__bf16)ap[grp * 16 + 8 + i];
    }
#pragma unroll
    for (int i = 0; i < 16; ++i)               // B: K = grp*16 + i
      bf[i] = (__bf16)bp[grp * 16 + i];
    acc = wmma_bf16(a, bf, acc);
  }

#pragma unroll
  for (int r = 0; r < 8; ++r) {                // C layout: M = r + 8*grp, N = ln
    const int tok = rowTile * 16 + r + 8 * grp;
    const int o   = ct * 16 + ln;
    const float v = acc[r] + bb[o];
    const int b = tok >> 11, s = tok & (S_ - 1);
    const int h = o >> 6,    d = o & 63;
    const size_t bh = (size_t)(b * H_ + h);
    if (mat == 0)      qb [(bh * S_ + s) * D_ + d] = (__bf16)v;
    else if (mat == 1) kbm[(bh * S_ + s) * D_ + d] = (__bf16)v;
    else               vt [(bh * D_ + d) * S_ + s] = (__bf16)v;
  }
}

// ---------------------------------------------------------------------------
// Kernel 2: flash attention with fused topological bias.
// Block = 8 waves sharing one (b,h); K/V k-tiles are staged in LDS by the
// Tensor Data Mover (double-buffered, issued by wave 0, TENSORcnt-tracked).
// Each wave owns a 16-row q-tile; online f32 softmax; P goes through LDS to
// re-shape C-layout -> A-fragment for the PV WMMA.
// ---------------------------------------------------------------------------
__global__ __launch_bounds__(256) void attn_kernel(
    const __bf16* __restrict__ qb, const __bf16* __restrict__ kbm,
    const __bf16* __restrict__ vt,
    const float* __restrict__ top, const float* __restrict__ topw,
    const float* __restrict__ topb, float* __restrict__ obuf)
{
  __shared__ __bf16 kst[2][32 * 64];           // K tile: 32 k-rows x 64 d (4 KB x2)
  __shared__ __bf16 vst[2][64 * 32];           // V^T tile: 64 d-rows x 32 k (4 KB x2)
  __shared__ __bf16 plds[8][16 * 32];          // per-wave P staging (8 KB)

  const int wave = threadIdx.x >> 5, lane = threadIdx.x & 31;
  const int grp = lane >> 4, ln = lane & 15;
  const int bh   = blockIdx.x >> 4;            // 0..15 = (b,h)
  const int qblk = blockIdx.x & 15;            // 16 blocks per (b,h)
  const int b = bh >> 3, h = bh & 7;
  const int qtile = qblk * 128 + wave * 16;    // this wave's q rows
  const __bf16* Q = qb  + (size_t)bh * S_ * D_;
  const __bf16* K = kbm + (size_t)bh * S_ * D_;
  const __bf16* V = vt  + (size_t)bh * D_ * S_;   // [d][s]

  // Q A-fragments (row m = ln, K-dim d = 64 -> two frags)
  const __bf16* qrowp = Q + (size_t)(qtile + ln) * D_;
  v16bf qf0 = ld8x2(qrowp + grp * 16,      qrowp + grp * 16 + 8);
  v16bf qf1 = ld8x2(qrowp + 32 + grp * 16, qrowp + 40 + grp * 16);

  float tw[16];
#pragma unroll
  for (int i = 0; i < 16; ++i) tw[i] = topw[i];
  const float tb = topb[0];

  float mrow[8], lrow[8];
  v8f o0 = {}, o1 = {}, o2 = {}, o3 = {};
#pragma unroll
  for (int r = 0; r < 8; ++r) { mrow[r] = -1e30f; lrow[r] = 0.0f; }

  __bf16* pl = plds[wave];
  const float scale = 0.125f;                  // 1/sqrt(64)

  // prologue: TDM-stage the first K/V tile into buffer 0
  if (wave == 0) {
    tdm_load_2d_bf16(lds_off(kst[0]), K, D_, S_, D_, 32, D_);
    tdm_load_2d_bf16(lds_off(vst[0]), V, S_, D_, 32, D_, S_);
  }

  for (int kt = 0; kt < S_; kt += 32) {
    const int cur = (kt >> 5) & 1, nxt = cur ^ 1;
    if (wave == 0) {
      if (kt + 32 < S_) {
        // stream next tile while this one is consumed
        tdm_load_2d_bf16(lds_off(kst[nxt]), K + (size_t)(kt + 32) * D_,
                         D_, S_, D_, 32, D_);
        tdm_load_2d_bf16(lds_off(vst[nxt]), V + (kt + 32),
                         S_, D_, 32, D_, S_);
        __builtin_amdgcn_s_wait_tensorcnt(2);  // retire the 'cur' pair
      } else {
        __builtin_amdgcn_s_wait_tensorcnt(0);
      }
    }
    __syncthreads();                           // LDS tile visible to all waves
    const __bf16* kl = kst[cur];
    const __bf16* vl = vst[cur];

    // ---- scores S = Q K^T : two 16x16 f32 C tiles (cols kt..+15, kt+16..+31)
    v8f s0 = {}, s1 = {};
    {
      const __bf16* k0p = kl + (size_t)ln * D_;         // B col n = ln
      const __bf16* k1p = kl + (size_t)(16 + ln) * D_;
      s0 = wmma_bf16(qf0, ld16(k0p + grp * 16), s0);    // d 0..31
      s1 = wmma_bf16(qf0, ld16(k1p + grp * 16), s1);
      s0 = wmma_bf16(qf1, ld16(k0p + 32 + grp * 16), s0);   // d 32..63
      s1 = wmma_bf16(qf1, ld16(k1p + 32 + grp * 16), s1);
    }

    // prefetch next bias tile (global_prefetch_b8)
    __builtin_prefetch(top + ((size_t)(b * S_ + qtile + ln) * S_ + kt + 32) * 16, 0, 0);

    // ---- fused topological bias: s = s*scale + (top[b,q,k,:] . tw + tb)
#pragma unroll
    for (int r = 0; r < 8; ++r) {
      const int qrow = qtile + r + 8 * grp;
      const size_t base = (size_t)(b * S_ + qrow) * S_;
      const float4* t0 = reinterpret_cast<const float4*>(top + (base + kt + ln) * 16);
      const float4* t1 = reinterpret_cast<const float4*>(top + (base + kt + 16 + ln) * 16);
      float b0 = tb, b1 = tb;
#pragma unroll
      for (int j = 0; j < 4; ++j) {
        float4 u = t0[j], v = t1[j];
        b0 += u.x * tw[4*j] + u.y * tw[4*j+1] + u.z * tw[4*j+2] + u.w * tw[4*j+3];
        b1 += v.x * tw[4*j] + v.y * tw[4*j+1] + v.z * tw[4*j+2] + v.w * tw[4*j+3];
      }
      s0[r] = s0[r] * scale + b0;
      s1[r] = s1[r] * scale + b1;
    }

    // ---- online softmax (row = 16-lane group; xor {1,2,4,8} reduces a row)
#pragma unroll
    for (int r = 0; r < 8; ++r) {
      float mx = fmaxf(s0[r], s1[r]);
      mx = fmaxf(mx, __shfl_xor(mx, 1));
      mx = fmaxf(mx, __shfl_xor(mx, 2));
      mx = fmaxf(mx, __shfl_xor(mx, 4));
      mx = fmaxf(mx, __shfl_xor(mx, 8));
      const float mnew  = fmaxf(mrow[r], mx);
      const float alpha = __expf(mrow[r] - mnew);
      mrow[r] = mnew;
      const float p0 = __expf(s0[r] - mnew);
      const float p1 = __expf(s1[r] - mnew);
      s0[r] = p0; s1[r] = p1;
      float rs = p0 + p1;
      rs += __shfl_xor(rs, 1);
      rs += __shfl_xor(rs, 2);
      rs += __shfl_xor(rs, 4);
      rs += __shfl_xor(rs, 8);
      lrow[r] = lrow[r] * alpha + rs;
      o0[r] *= alpha; o1[r] *= alpha; o2[r] *= alpha; o3[r] *= alpha;
    }

    // ---- P (C layout) -> LDS row-major 16x32 -> reload as A fragment
#pragma unroll
    for (int r = 0; r < 8; ++r) {
      pl[(r + 8 * grp) * 32 + ln]      = (__bf16)s0[r];
      pl[(r + 8 * grp) * 32 + 16 + ln] = (__bf16)s1[r];
    }
    // same-wave DS RAW: compiler inserts s_wait_dscnt
    v16bf pa = ld8x2(pl + ln * 32 + grp * 16, pl + ln * 32 + grp * 16 + 8);

    // ---- O += P V   (V^T LDS tile: row d, 32 contiguous k)
    o0 = wmma_bf16(pa, ld16(vl + (size_t)(0 * 16 + ln) * 32 + grp * 16), o0);
    o1 = wmma_bf16(pa, ld16(vl + (size_t)(1 * 16 + ln) * 32 + grp * 16), o1);
    o2 = wmma_bf16(pa, ld16(vl + (size_t)(2 * 16 + ln) * 32 + grp * 16), o2);
    o3 = wmma_bf16(pa, ld16(vl + (size_t)(3 * 16 + ln) * 32 + grp * 16), o3);

    __syncthreads();                           // all reads done before reuse
  }

  // ---- epilogue: O /= l, scatter to [b,s,h*64+d] f32 for the output proj
#pragma unroll
  for (int r = 0; r < 8; ++r) {
    const int qrow = qtile + r + 8 * grp;
    const float inv = 1.0f / lrow[r];
    float* orow = obuf + (size_t)(b * S_ + qrow) * DM_ + h * D_;
    orow[0 * 16 + ln] = o0[r] * inv;
    orow[1 * 16 + ln] = o1[r] * inv;
    orow[2 * 16 + ln] = o2[r] * inv;
    orow[3 * 16 + ln] = o3[r] * inv;
  }
}

// ---------------------------------------------------------------------------
// Kernel 3: output projection  out = O @ wo^T + bo   (f32 out)
// ---------------------------------------------------------------------------
__global__ __launch_bounds__(256) void out_proj_kernel(
    const float* __restrict__ xin, const float* __restrict__ wo,
    const float* __restrict__ bo, float* __restrict__ out)
{
  const int wave = threadIdx.x >> 5, lane = threadIdx.x & 31;
  const int grp = lane >> 4, ln = lane & 15;
  const int tile = blockIdx.x * 8 + wave;      // 0 .. 8191
  const int rowTile = tile >> 5;               // 256 row tiles
  const int colTile = tile & 31;               // 32 col tiles
  const int arow = rowTile * 16 + ln;
  const int bcol = colTile * 16 + ln;

  v8f acc = {};
  for (int kt = 0; kt < DM_; kt += 32) {
    v16bf a, bf;
    const float* ap = xin + (size_t)arow * DM_ + kt;
    const float* bp = wo  + (size_t)bcol * DM_ + kt;
#pragma unroll
    for (int i = 0; i < 8; ++i) {
      a[i]     = (__bf16)ap[grp * 16 + i];
      a[i + 8] = (__bf16)ap[grp * 16 + 8 + i];
    }
#pragma unroll
    for (int i = 0; i < 16; ++i) bf[i] = (__bf16)bp[grp * 16 + i];
    acc = wmma_bf16(a, bf, acc);
  }
#pragma unroll
  for (int r = 0; r < 8; ++r) {
    const int tok = rowTile * 16 + r + 8 * grp;
    const int o   = colTile * 16 + ln;
    out[(size_t)tok * DM_ + o] = acc[r] + bo[o];
  }
}

// ---------------------------------------------------------------------------
// Host launcher
// ---------------------------------------------------------------------------
extern "C" void kernel_launch(void* const* d_in, const int* in_sizes, int n_in,
                              void* d_out, int out_size, void* d_ws, size_t ws_size,
                              hipStream_t stream) {
  (void)in_sizes; (void)n_in; (void)out_size; (void)ws_size;
  const float* input = (const float*)d_in[0];
  const float* top   = (const float*)d_in[1];
  const float* top_w = (const float*)d_in[2];
  const float* top_b = (const float*)d_in[3];
  const float* wq    = (const float*)d_in[4];
  const float* bq    = (const float*)d_in[5];
  const float* wk    = (const float*)d_in[6];
  const float* bk    = (const float*)d_in[7];
  const float* wv    = (const float*)d_in[8];
  const float* bv    = (const float*)d_in[9];
  const float* wo    = (const float*)d_in[10];
  const float* bo    = (const float*)d_in[11];

  // workspace layout (needs 20 MiB):
  //   [0,4M)    Q  bf16 [b,h,s,d]
  //   [4M,8M)   K  bf16 [b,h,s,d]
  //   [8M,12M)  V  bf16 [b,h,d,s]  (transposed)
  //   [12M,20M) O  f32  [b,s,h*d]
  char* ws = (char*)d_ws;
  __bf16* qb   = (__bf16*)(ws);
  __bf16* kbm  = (__bf16*)(ws + ((size_t)4  << 20));
  __bf16* vt   = (__bf16*)(ws + ((size_t)8  << 20));
  float*  obuf = (float*) (ws + ((size_t)12 << 20));

  // K1: 24576 tiles / 8 waves = 3072 blocks
  qkv_proj_kernel<<<3072, 256, 0, stream>>>(input, wq, bq, wk, bk, wv, bv,
                                            qb, kbm, vt);
  // K2: 16 (b,h) x 16 q-blocks = 256 blocks
  attn_kernel<<<256, 256, 0, stream>>>(qb, kbm, vt, top, top_w, top_b, obuf);
  // K3: 8192 tiles / 8 waves = 1024 blocks
  out_proj_kernel<<<1024, 256, 0, stream>>>(obuf, wo, bo, (float*)d_out);
}
